// GeneMambaV0_1_27462020890877
// MI455X (gfx1250) — compile-verified, matrix operand order checked by reading, not run on hardware
//
#include <hip/hip_runtime.h>
#include <math.h>

// ---------------- model constants ----------------
constexpr int Bn = 16, Ln = 4096, Hn = 128;
constexpr int DI = 256, DSn = 16, NH = 4, HD = 64, CH = 64, NCH = Ln / CH;
constexpr int DIP = 2 * DI + 2 * DSn + NH;      // 548
constexpr int DIP_PAD = 576;                    // padded to multiple of 32
constexpr int CDIM = DI + 2 * DSn;              // 288
constexpr int DCONV = 4, LAT = 64;
constexpr float EPSF = 1e-5f;
constexpr size_t NBL = (size_t)Bn * Ln;         // 65536 rows

// ---------------- workspace layout (float units) ----------------
constexpr size_t SZ_H   = NBL * Hn;             // 8,388,608
constexpr size_t SZ_ZX  = NBL * DIP;
constexpr size_t SZ_XBC = NBL * CDIM;
constexpr size_t SZ_DT  = NBL * NH;
constexpr size_t SZ_Y   = NBL * DI;
constexpr size_t SZ_ST  = (size_t)Bn * NCH * NH * DSn * HD;
constexpr size_t SZ_DAS = (size_t)Bn * NCH * NH;
constexpr size_t SZ_ABF = NBL * DI / 2;         // bf16 A buffer (NBL*256 bf16)
constexpr size_t SZ_WBF = 65536;                // bf16 weight buffer (128K bf16)

constexpr size_t O_XF  = 0;
constexpr size_t O_XB  = O_XF  + SZ_H;
constexpr size_t O_T0  = O_XB  + SZ_H;
constexpr size_t O_T1  = O_T0  + SZ_H;
constexpr size_t O_ZX  = O_T1  + SZ_H;
constexpr size_t O_XBC = O_ZX  + SZ_ZX;
constexpr size_t O_DT  = O_XBC + SZ_XBC;
constexpr size_t O_Y   = O_DT  + SZ_DT;
constexpr size_t O_NRM = O_Y   + SZ_Y;
constexpr size_t O_ST  = O_NRM + SZ_Y;
constexpr size_t O_STP = O_ST  + SZ_ST;
constexpr size_t O_DAS = O_STP + SZ_ST;
constexpr size_t O_CAT = O_DAS + SZ_DAS;
constexpr size_t O_FUS = O_CAT + SZ_Y;
constexpr size_t O_ABF = O_FUS + SZ_H;
constexpr size_t O_WBF = O_ABF + SZ_ABF;

// ---------------- bf16 helpers / WMMA types ----------------
typedef __bf16 v4bf  __attribute__((ext_vector_type(4)));
typedef __bf16 v8bf  __attribute__((ext_vector_type(8)));
typedef __bf16 v16bf __attribute__((ext_vector_type(16)));
typedef float  v8f   __attribute__((ext_vector_type(8)));

__device__ __forceinline__ __bf16 f2bf(float f) {
    unsigned u = __builtin_bit_cast(unsigned, f);
    unsigned r = u + 0x7fffu + ((u >> 16) & 1u);      // round-to-nearest-even
    unsigned short hs = (unsigned short)(r >> 16);
    return __builtin_bit_cast(__bf16, hs);
}
__device__ __forceinline__ float sigf(float x) { return 1.0f / (1.0f + __expf(-x)); }

// Build a 16x32 operand fragment for one lane from a bf16 row:
// elements 0..7  <- K = k0 + 8*half .. +7      (one 16B load)
// elements 8..15 <- K = k0 + 16 + 8*half .. +7 (one 16B load)
__device__ __forceinline__ v16bf load_frag(const __bf16* __restrict__ row,
                                           int k0, int half) {
    const v8bf lo = *(const v8bf*)(row + k0 + half * 8);
    const v8bf hi = *(const v8bf*)(row + k0 + 16 + half * 8);
    return __builtin_shufflevector(lo, hi, 0, 1, 2, 3, 4, 5, 6, 7,
                                           8, 9, 10, 11, 12, 13, 14, 15);
}

// ============================================================
// WMMA GEMM: C[M,N] = A[M,K] * W[Npad,K]^T (+bias, +activation)
// A, W pre-converted to bf16 (W zero-padded to Npad mult. of 32).
// block = 128 threads (4 waves); each wave computes a 16x32 C tile
// (one A fragment reused over two B fragments). K fully unrolled.
// act: 0 = none, 1 = exact GELU.
// ============================================================
template <int K>
__global__ void gemm_bf16_wmma(const __bf16* __restrict__ A,
                               const __bf16* __restrict__ W,
                               float* __restrict__ C,
                               const float* __restrict__ bias,
                               int N, int act) {
    const int lane = threadIdx.x & 31;
    const int wave = threadIdx.x >> 5;
    const int tileM = blockIdx.x * 64 + wave * 16;
    const int tileN = blockIdx.y * 32;
    const int half = lane >> 4;
    const int r    = lane & 15;

    const __bf16* arow  = A + (size_t)(tileM + r) * K;
    const __bf16* brow0 = W + (size_t)(tileN + r) * K;
    const __bf16* brow1 = W + (size_t)(tileN + 16 + r) * K;

    v8f acc0 = {}, acc1 = {};
#pragma unroll
    for (int k0 = 0; k0 < K; k0 += 32) {
        const v16bf af  = load_frag(arow,  k0, half);
        const v16bf bf0 = load_frag(brow0, k0, half);
        const v16bf bf1 = load_frag(brow1, k0, half);
        acc0 = __builtin_amdgcn_wmma_f32_16x16x32_bf16(
            false, af, false, bf0, (short)0, acc0, false, false);
        acc1 = __builtin_amdgcn_wmma_f32_16x16x32_bf16(
            false, af, false, bf1, (short)0, acc1, false, false);
    }

#pragma unroll
    for (int tt = 0; tt < 2; ++tt) {
        const int col = tileN + tt * 16 + r;
        if (col < N) {
            const float bv = bias ? bias[col] : 0.0f;
            const v8f acc = tt ? acc1 : acc0;
#pragma unroll
            for (int j = 0; j < 8; ++j) {
                const int row = tileM + half * 8 + j;   // ISA C/D layout
                float v = acc[j] + bv;
                if (act == 1) v = 0.5f * v * (1.0f + erff(v * 0.70710678118f));
                C[(size_t)row * N + col] = v;
            }
        }
    }
}

// ---------------- fp32 -> bf16 converters ----------------
__global__ void cvt_a_kernel(const float* __restrict__ src,
                             __bf16* __restrict__ dst, size_t n4) {
    size_t i = (size_t)blockIdx.x * blockDim.x + threadIdx.x;
    if (i >= n4) return;
    const float4 v = ((const float4*)src)[i];
    v4bf o;
    o[0] = f2bf(v.x); o[1] = f2bf(v.y); o[2] = f2bf(v.z); o[3] = f2bf(v.w);
    ((v4bf*)dst)[i] = o;
}

__global__ void cvt_w_kernel(const float* __restrict__ W,
                             __bf16* __restrict__ Wb, int N, int K, int Npad) {
    size_t idx = (size_t)blockIdx.x * blockDim.x + threadIdx.x;
    if (idx >= (size_t)Npad * K) return;
    const int n = (int)(idx / K);
    const int k = (int)(idx % K);
    Wb[idx] = (n < N) ? f2bf(W[(size_t)n * K + k])
                      : __builtin_bit_cast(__bf16, (unsigned short)0);
}

// ============================================================
// Elementwise kernels
// ============================================================
__global__ void embed_kernel(const float* __restrict__ expr,
                             const float* __restrict__ ew,
                             const float* __restrict__ eb,
                             float* __restrict__ xf, float* __restrict__ xb) {
    size_t idx = (size_t)blockIdx.x * blockDim.x + threadIdx.x;
    if (idx >= NBL * Hn) return;
    int h = idx % Hn;
    int g = (idx / Hn) % Ln;
    int b = idx / ((size_t)Hn * Ln);
    float v = expr[(size_t)b * Ln + g] * ew[h] + eb[h];
    xf[idx] = v;
    xb[(((size_t)b * Ln + (Ln - 1 - g)) * Hn) + h] = v;
}

__global__ void reverse_kernel(const float* __restrict__ in, float* __restrict__ out) {
    size_t idx = (size_t)blockIdx.x * blockDim.x + threadIdx.x;
    if (idx >= NBL * Hn) return;
    int h = idx % Hn;
    int l = (idx / Hn) % Ln;
    int b = idx / ((size_t)Hn * Ln);
    out[(((size_t)b * Ln + (Ln - 1 - l)) * Hn) + h] = in[idx];
}

__global__ void concat_kernel(const float* __restrict__ fw, const float* __restrict__ bw,
                              float* __restrict__ cat) {
    size_t idx = (size_t)blockIdx.x * blockDim.x + threadIdx.x;
    if (idx >= NBL * 2 * Hn) return;
    int c = idx % (2 * Hn);
    size_t row = idx / (2 * Hn);
    cat[idx] = (c < Hn) ? fw[row * Hn + c] : bw[row * Hn + (c - Hn)];
}

// depthwise causal conv over L (pad-left 3) + bias + SiLU
__global__ void conv_kernel(const float* __restrict__ zx,
                            const float* __restrict__ cw,
                            const float* __restrict__ cb,
                            float* __restrict__ xbc) {
    size_t idx = (size_t)blockIdx.x * blockDim.x + threadIdx.x;
    if (idx >= NBL * CDIM) return;
    int ch = idx % CDIM;
    int l  = (idx / CDIM) % Ln;
    int b  = idx / ((size_t)CDIM * Ln);
    float acc = cb[ch];
#pragma unroll
    for (int k = 0; k < DCONV; ++k) {
        int ll = l - (DCONV - 1) + k;
        if (ll >= 0) acc += cw[ch * DCONV + k] * zx[((size_t)b * Ln + ll) * DIP + DI + ch];
    }
    xbc[idx] = acc * sigf(acc);
}

__global__ void dt_kernel(const float* __restrict__ zx,
                          const float* __restrict__ dtb,
                          float* __restrict__ dt) {
    size_t idx = (size_t)blockIdx.x * blockDim.x + threadIdx.x;
    if (idx >= NBL * NH) return;
    int h = idx % NH;
    size_t row = idx / NH;
    float x = zx[row * DIP + DI + CDIM + h] + dtb[h];
    dt[idx] = (x > 20.0f) ? x : log1pf(__expf(x));
}

// ============================================================
// SSD chunk kernel: one block of 64 threads per (b, chunk, head).
// ============================================================
__global__ void ssd_chunk_kernel(const float* __restrict__ xbc,
                                 const float* __restrict__ dt,
                                 const float* __restrict__ Alog,
                                 float* __restrict__ y,
                                 float* __restrict__ st,
                                 float* __restrict__ dAs) {
    const int blk = blockIdx.x;                 // (b*NCH + c)*NH + h
    const int h = blk % NH;
    const int c = (blk / NH) % NCH;
    const int b = blk / (NH * NCH);
    const int l = threadIdx.x;                  // 0..63 (row in chunk)

    __shared__ float xs[CH][HD + 1];
    __shared__ float Bs[CH][DSn];
    __shared__ float Cs[CH][DSn];
    __shared__ float av[CH];
    __shared__ float acs[CH];

    const size_t base = (size_t)b * Ln + (size_t)c * CH + l;
    const float dtl = dt[base * NH + h];
    const float A = -__expf(Alog[h]);
    av[l] = A * dtl;
    const float* xrow = xbc + base * CDIM + h * HD;
#pragma unroll 4
    for (int p = 0; p < HD; ++p) xs[l][p] = xrow[p] * dtl;
    const float* brow = xbc + base * CDIM + DI;
#pragma unroll
    for (int n = 0; n < DSn; ++n) { Bs[l][n] = brow[n]; Cs[l][n] = brow[DSn + n]; }
    __syncthreads();

    float s = 0.f;
    for (int j = 0; j <= l; ++j) s += av[j];
    acs[l] = s;
    __syncthreads();

    // Y_diag row l
    float yd[HD];
#pragma unroll 4
    for (int p = 0; p < HD; ++p) yd[p] = 0.f;
    for (int ss = 0; ss <= l; ++ss) {
        float d = 0.f;
#pragma unroll
        for (int n = 0; n < DSn; ++n) d += Cs[l][n] * Bs[ss][n];
        const float w = d * __expf(acs[l] - acs[ss]);
#pragma unroll 4
        for (int p = 0; p < HD; ++p) yd[p] += w * xs[ss][p];
    }
    float* yrow = y + base * DI + h * HD;
#pragma unroll 4
    for (int p = 0; p < HD; ++p) yrow[p] = yd[p];

    // chunk state: state[n][p] = sum_l B[l][n] * exp(acs[63]-acs[l]) * xdt[l][p]
    const int n  = l & 15;
    const int p0 = (l >> 4) * 16;
    const float fin = acs[CH - 1];
    float stv[16];
#pragma unroll
    for (int k = 0; k < 16; ++k) stv[k] = 0.f;
    for (int j = 0; j < CH; ++j) {
        const float w = Bs[j][n] * __expf(fin - acs[j]);
#pragma unroll
        for (int k = 0; k < 16; ++k) stv[k] += w * xs[j][p0 + k];
    }
    float* sp = st + (size_t)blk * (DSn * HD);
#pragma unroll
    for (int k = 0; k < 16; ++k) sp[n * HD + p0 + k] = stv[k];
    if (l == 0) dAs[blk] = fin;
}

// inter-chunk scan: one block of 256 threads per (b,h); serial over 64 chunks
__global__ void ssd_scan_kernel(const float* __restrict__ st,
                                const float* __restrict__ dAs,
                                float* __restrict__ stp) {
    const int b = blockIdx.x / NH;
    const int h = blockIdx.x % NH;
    const int e0 = threadIdx.x * 4;
    float S[4] = {0.f, 0.f, 0.f, 0.f};
    for (int c = 0; c < NCH; ++c) {
        const size_t blk = ((size_t)b * NCH + c) * NH + h;
        const size_t idx = blk * (DSn * HD) + e0;
#pragma unroll
        for (int k = 0; k < 4; ++k) stp[idx + k] = S[k];
        const float dec = __expf(dAs[blk]);
#pragma unroll
        for (int k = 0; k < 4; ++k) S[k] = dec * S[k] + st[idx + k];
    }
}

// Y_off + D skip: y += exp(acs[l]) * C @ state_prev + x*D
__global__ void ssd_yoff_kernel(const float* __restrict__ xbc,
                                const float* __restrict__ dt,
                                const float* __restrict__ Alog,
                                const float* __restrict__ Dp,
                                const float* __restrict__ stp,
                                float* __restrict__ y) {
    const int blk = blockIdx.x;
    const int h = blk % NH;
    const int c = (blk / NH) % NCH;
    const int b = blk / (NH * NCH);
    const int l = threadIdx.x;

    __shared__ float Cs[CH][DSn];
    __shared__ float av[CH];
    __shared__ float spl[DSn * HD];

    const size_t base = (size_t)b * Ln + (size_t)c * CH + l;
    const float dtl = dt[base * NH + h];
    const float A = -__expf(Alog[h]);
    av[l] = A * dtl;
    const float* brow = xbc + base * CDIM + DI;
#pragma unroll
    for (int n = 0; n < DSn; ++n) Cs[l][n] = brow[DSn + n];
    const size_t sbase = (size_t)blk * (DSn * HD);
#pragma unroll
    for (int e = 0; e < 16; ++e) spl[l * 16 + e] = stp[sbase + l * 16 + e];
    __syncthreads();

    float s = 0.f;
    for (int j = 0; j <= l; ++j) s += av[j];
    const float sc = __expf(s);
    const float dp = Dp[h];
    const float* xrow = xbc + base * CDIM + h * HD;
    float* yrow = y + base * DI + h * HD;
#pragma unroll 4
    for (int p = 0; p < HD; ++p) {
        float acc = 0.f;
#pragma unroll
        for (int n = 0; n < DSn; ++n) acc += Cs[l][n] * spl[n * HD + p];
        yrow[p] += sc * acc + xrow[p] * dp;
    }
}

// gated RMS-norm over D_INNER: one block of 256 threads per token
__global__ void gate_norm_kernel(const float* __restrict__ y,
                                 const float* __restrict__ zx,
                                 const float* __restrict__ nw,
                                 float* __restrict__ nrm) {
    const size_t row = blockIdx.x;
    const int d = threadIdx.x;
    __shared__ float red[DI];
    const float z = zx[row * DIP + d];
    const float yf = y[row * DI + d] * (z * sigf(z));
    red[d] = yf * yf;
    __syncthreads();
    for (int s = DI / 2; s > 0; s >>= 1) {
        if (d < s) red[d] += red[d + s];
        __syncthreads();
    }
    const float inv = rsqrtf(red[0] / (float)DI + EPSF);
    nrm[row * DI + d] = yf * inv * nw[d];
}

// ============================================================
// Head: lm / lv / sample / gate / pred ; one block of 128 per token
// d_out layout: pred (B*G) | lm (B*G*64) | lv (B*G*64) | sample (B*G*64)
// ============================================================
__global__ void head_kernel(const float* __restrict__ fus,
                            const float* __restrict__ noise,
                            const float* __restrict__ lm_w, const float* __restrict__ lm_b,
                            const float* __restrict__ lv_w, const float* __restrict__ lv_b,
                            const float* __restrict__ gate_w, const float* __restrict__ gate_b,
                            const float* __restrict__ outp_w, const float* __restrict__ outp_b,
                            const float* __restrict__ reg_gate,
                            float* __restrict__ out) {
    const size_t row = blockIdx.x;
    const int t = threadIdx.x;
    __shared__ float fr[Hn];
    __shared__ float samp[LAT];
    __shared__ float red[Hn];

    fr[t] = fus[row * Hn + t];
    __syncthreads();

    const size_t OP = NBL;         // offset past pred block
    if (t < LAT) {
        float sm = lm_b[t], sv = lv_b[t];
        for (int j = 0; j < Hn; ++j) {
            const float f = fr[j];
            sm += lm_w[t * Hn + j] * f;
            sv += lv_w[t * Hn + j] * f;
        }
        const float nz = noise[row * LAT + t];
        const float sa = sm + nz * __expf(0.5f * sv);
        samp[t] = sa;
        out[OP + row * LAT + t] = sm;
        out[OP + NBL * LAT + row * LAT + t] = sv;
        out[OP + (size_t)2 * NBL * LAT + row * LAT + t] = sa;
    }
    __syncthreads();

    float g = gate_b[t];
    for (int j = 0; j < LAT; ++j) g += gate_w[t * LAT + j] * samp[j];
    const float hv = fr[t] * sigf(g);
    red[t] = hv * outp_w[t];
    __syncthreads();
    for (int s = Hn / 2; s > 0; s >>= 1) {
        if (t < s) red[t] += red[t + s];
        __syncthreads();
    }
    if (t == 0) {
        const int gidx = (int)(row % Ln);
        out[row] = (red[0] + outp_b[0]) * sigf(reg_gate[gidx]);
    }
}

// ============================================================
// Host-side helpers
// ============================================================
static void run_gemm128(const float* A, const float* W, float* C, const float* bias,
                        int N, int Npad, int act, float* ws, hipStream_t stream) {
    __bf16* abf = (__bf16*)(ws + O_ABF);
    __bf16* wbf = (__bf16*)(ws + O_WBF);
    size_t n4 = NBL * 128 / 4;
    cvt_a_kernel<<<(unsigned)((n4 + 255) / 256), 256, 0, stream>>>(A, abf, n4);
    size_t nw = (size_t)Npad * 128;
    cvt_w_kernel<<<(unsigned)((nw + 255) / 256), 256, 0, stream>>>(W, wbf, N, 128, Npad);
    dim3 grid((unsigned)(NBL / 64), Npad / 32);
    gemm_bf16_wmma<128><<<grid, 128, 0, stream>>>(abf, wbf, C, bias, N, act);
}

static void run_gemm256(const float* A, const float* W, float* C, const float* bias,
                        int N, int Npad, int act, float* ws, hipStream_t stream) {
    __bf16* abf = (__bf16*)(ws + O_ABF);
    __bf16* wbf = (__bf16*)(ws + O_WBF);
    size_t n4 = NBL * 256 / 4;
    cvt_a_kernel<<<(unsigned)((n4 + 255) / 256), 256, 0, stream>>>(A, abf, n4);
    size_t nw = (size_t)Npad * 256;
    cvt_w_kernel<<<(unsigned)((nw + 255) / 256), 256, 0, stream>>>(W, wbf, N, 256, Npad);
    dim3 grid((unsigned)(NBL / 64), Npad / 32);
    gemm_bf16_wmma<256><<<grid, 128, 0, stream>>>(abf, wbf, C, bias, N, act);
}

static void run_mamba_layer(int layer,
                            const float* u_in, float* u_out, float* ws,
                            const float* in_proj_w, const float* conv_w, const float* conv_b,
                            const float* dt_bias, const float* A_log, const float* D_par,
                            const float* norm_w, const float* out_proj_w,
                            hipStream_t stream) {
    float* zx  = ws + O_ZX;
    float* xbc = ws + O_XBC;
    float* dtb = ws + O_DT;
    float* yb  = ws + O_Y;
    float* nrm = ws + O_NRM;
    float* st  = ws + O_ST;
    float* stp = ws + O_STP;
    float* dAs = ws + O_DAS;

    const float* Wi = in_proj_w + (size_t)layer * DIP * Hn;
    const float* cw = conv_w + (size_t)layer * CDIM * DCONV;
    const float* cb = conv_b + (size_t)layer * CDIM;
    const float* db = dt_bias + (size_t)layer * NH;
    const float* Al = A_log + (size_t)layer * NH;
    const float* Dp = D_par + (size_t)layer * NH;
    const float* nw = norm_w + (size_t)layer * DI;
    const float* Wo = out_proj_w + (size_t)layer * Hn * DI;

    // in_proj: [NBL,128] @ [548,128]^T
    run_gemm128(u_in, Wi, zx, nullptr, DIP, DIP_PAD, 0, ws, stream);
    {
        size_t n = NBL * CDIM;
        conv_kernel<<<(unsigned)((n + 255) / 256), 256, 0, stream>>>(zx, cw, cb, xbc);
    }
    {
        size_t n = NBL * NH;
        dt_kernel<<<(unsigned)((n + 255) / 256), 256, 0, stream>>>(zx, db, dtb);
    }
    {
        unsigned nblk = (unsigned)((size_t)Bn * NCH * NH);
        ssd_chunk_kernel<<<nblk, CH, 0, stream>>>(xbc, dtb, Al, yb, st, dAs);
        ssd_scan_kernel<<<Bn * NH, 256, 0, stream>>>(st, dAs, stp);
        ssd_yoff_kernel<<<nblk, CH, 0, stream>>>(xbc, dtb, Al, Dp, stp, yb);
    }
    gate_norm_kernel<<<(unsigned)NBL, DI, 0, stream>>>(yb, zx, nw, nrm);
    // out_proj: [NBL,256] @ [128,256]^T
    run_gemm256(nrm, Wo, u_out, nullptr, Hn, Hn, 0, ws, stream);
}

extern "C" void kernel_launch(void* const* d_in, const int* in_sizes, int n_in,
                              void* d_out, int out_size, void* d_ws, size_t ws_size,
                              hipStream_t stream) {
    (void)in_sizes; (void)n_in; (void)out_size; (void)ws_size;
    const float* expression = (const float*)d_in[0];
    const float* noise      = (const float*)d_in[1];
    const float* expr_w     = (const float*)d_in[2];
    const float* expr_b     = (const float*)d_in[3];
    const float* in_proj_w  = (const float*)d_in[4];
    const float* conv_w     = (const float*)d_in[5];
    const float* conv_b     = (const float*)d_in[6];
    const float* dt_bias    = (const float*)d_in[7];
    const float* A_log      = (const float*)d_in[8];
    const float* D_par      = (const float*)d_in[9];
    const float* norm_w     = (const float*)d_in[10];
    const float* out_proj_w = (const float*)d_in[11];
    const float* fusion_w   = (const float*)d_in[12];
    const float* fusion_b   = (const float*)d_in[13];
    const float* lm_w       = (const float*)d_in[14];
    const float* lm_b       = (const float*)d_in[15];
    const float* lv_w       = (const float*)d_in[16];
    const float* lv_b       = (const float*)d_in[17];
    const float* gate_w     = (const float*)d_in[18];
    const float* gate_b     = (const float*)d_in[19];
    const float* outp_w     = (const float*)d_in[20];
    const float* outp_b     = (const float*)d_in[21];
    const float* reg_gate   = (const float*)d_in[22];

    float* ws = (float*)d_ws;
    float* xf  = ws + O_XF;
    float* xb  = ws + O_XB;
    float* t0  = ws + O_T0;
    float* t1  = ws + O_T1;
    float* cat = ws + O_CAT;
    float* fus = ws + O_FUS;
    float* out = (float*)d_out;

    // embed (+ write pre-reversed copy for backward branch)
    {
        size_t n = NBL * Hn;
        embed_kernel<<<(unsigned)((n + 255) / 256), 256, 0, stream>>>(
            expression, expr_w, expr_b, xf, xb);
    }

    // forward branch: layers 0,1
    run_mamba_layer(0, xf, t0, ws, in_proj_w, conv_w, conv_b, dt_bias, A_log,
                    D_par, norm_w, out_proj_w, stream);
    run_mamba_layer(1, t0, xf, ws, in_proj_w, conv_w, conv_b, dt_bias, A_log,
                    D_par, norm_w, out_proj_w, stream);
    // backward branch: layers 2,3 on reversed input, then un-reverse
    run_mamba_layer(2, xb, t0, ws, in_proj_w, conv_w, conv_b, dt_bias, A_log,
                    D_par, norm_w, out_proj_w, stream);
    run_mamba_layer(3, t0, t1, ws, in_proj_w, conv_w, conv_b, dt_bias, A_log,
                    D_par, norm_w, out_proj_w, stream);
    {
        size_t n = NBL * Hn;
        reverse_kernel<<<(unsigned)((n + 255) / 256), 256, 0, stream>>>(t1, t0);
    }

    // fusion: gelu(concat(fwd,bwd) @ Wf^T + bf)
    {
        size_t n = NBL * 2 * Hn;
        concat_kernel<<<(unsigned)((n + 255) / 256), 256, 0, stream>>>(xf, t0, cat);
        run_gemm256(cat, fusion_w, fus, fusion_b, Hn, Hn, 1, ws, stream);
    }

    // head
    head_kernel<<<(unsigned)NBL, Hn, 0, stream>>>(
        fus, noise, lm_w, lm_b, lv_w, lv_b, gate_w, gate_b,
        outp_w, outp_b, reg_gate, out);
}
